// BiLSTM2D_49297634623822
// MI455X (gfx1250) — matrix operand order repaired and checked by reading, not compiled, not run on hardware
//
#include <hip/hip_runtime.h>

#define B_   64
#define HH_  48
#define WW_  48
#define C_   192
#define HD_  48
#define M_   (B_*HH_*WW_)     /* 147456 spatial sites */
#define SP_  (HH_*WW_)        /* 2304 */
#define NSEQ (B_*WW_)         /* 3072 sequences per direction */
#define NG_  (4*HD_)          /* 192 gate columns per direction */
#define NDIR 4
#define NGALL (NDIR*NG_)      /* 768 */

typedef __attribute__((ext_vector_type(16))) __bf16 v16bf;
typedef __attribute__((ext_vector_type(8)))  float  v8f;

union FragBF { v16bf v; unsigned short u[16]; unsigned int d[8]; };

__device__ __forceinline__ unsigned short f2bf(float f) {
  union { float f; unsigned u; } x; x.f = f;
  unsigned r = x.u + 0x7FFFu + ((x.u >> 16) & 1u);   // round-to-nearest-even
  return (unsigned short)(r >> 16);
}

__device__ __forceinline__ float sigm(float v) { return 1.0f / (1.0f + __expf(-v)); }

// A fragment: 16x32 bf16 tile, row-major source, leading dim `ld` (ushorts).
// Lane<16: row M=lane, K 0..15; lane>=16: K 16..31.
// Two contiguous 16B runs per lane -> compiler emits 2x ds_load_b128.
__device__ __forceinline__ v16bf load_a_frag(const unsigned short* src, int ld, int lane) {
  FragBF f;
  const unsigned short* row = src + (lane & 15) * ld + (lane >> 4) * 16;
#pragma unroll
  for (int v = 0; v < 4; ++v) {
    f.d[v]     = *(const unsigned int*)(row + 2 * v);
    f.d[v + 4] = *(const unsigned int*)(row + 8 + 2 * v);
  }
  return f.v;
}

// B fragment from N-major (transposed) storage: src is [N][ld] row-major, ld>=32.
// Lane holds column n = lane&15; K = (lane>=16 ? 16 : 0) + 0..15 contiguous
// -> one 32B contiguous run per lane -> 2x b128 loads, no scalar gathers.
__device__ __forceinline__ v16bf load_b_frag_t(const unsigned short* src, int ld, int lane) {
  FragBF f;
  const unsigned short* row = src + (lane & 15) * ld + (lane >> 4) * 16;
#pragma unroll
  for (int v = 0; v < 8; ++v)
    f.d[v] = *(const unsigned int*)(row + 2 * v);
  return f.v;
}

// ---------------------------------------------------------------------------
// Pack weights to bf16. All B matrices are kept N-major (natural PyTorch
// layout), so packing is a pure elementwise conversion; K of Whh padded to 64.
// ---------------------------------------------------------------------------
__global__ void pack_params(
    const float* __restrict__ vwf, const float* __restrict__ vhf,
    const float* __restrict__ vbif, const float* __restrict__ vbhf,
    const float* __restrict__ vwb, const float* __restrict__ vhb,
    const float* __restrict__ vbib, const float* __restrict__ vbhb,
    const float* __restrict__ hwf, const float* __restrict__ hhf,
    const float* __restrict__ hbif, const float* __restrict__ hbhf,
    const float* __restrict__ hwb, const float* __restrict__ hhb,
    const float* __restrict__ hbib, const float* __restrict__ hbhb,
    const float* __restrict__ projw,
    unsigned short* __restrict__ wihT,    // [NGALL][C_]      N x K
    unsigned short* __restrict__ whhT,    // [NDIR][NG_][64]  N x K, K padded 48->64
    unsigned short* __restrict__ projT,   // [C_][NG_]        N x K
    float* __restrict__ biasP)            // [NGALL] b_ih + b_hh
{
  const float* wih[4] = {vwf, vwb, hwf, hwb};
  const float* whh[4] = {vhf, vhb, hhf, hhb};
  const float* bih[4] = {vbif, vbib, hbif, hbib};
  const float* bhh[4] = {vbhf, vbhb, hbhf, hbhb};
  int tid = blockIdx.x * blockDim.x + threadIdx.x;
  int stride = gridDim.x * blockDim.x;
  for (int i = tid; i < NGALL * C_; i += stride) {
    int dir = i / (NG_ * C_), rem = i % (NG_ * C_);
    wihT[i] = f2bf(wih[dir][rem]);
  }
  for (int i = tid; i < NDIR * NG_ * 64; i += stride) {
    int dir = i / (NG_ * 64), rem = i % (NG_ * 64), g = rem / 64, k = rem % 64;
    whhT[i] = (k < HD_) ? f2bf(whh[dir][g * HD_ + k]) : (unsigned short)0;
  }
  for (int i = tid; i < C_ * NG_; i += stride)
    projT[i] = f2bf(projw[i]);
  for (int i = tid; i < NGALL; i += stride) {
    int dir = i / NG_, g = i % NG_;
    biasP[i] = bih[dir][g] + bhh[dir][g];
  }
}

__global__ void convert_x(const float* __restrict__ x, unsigned short* __restrict__ o, int n) {
  int i = blockIdx.x * blockDim.x + threadIdx.x;
  int stride = gridDim.x * blockDim.x;
  for (; i < n; i += stride) o[i] = f2bf(x[i]);
}

// ---------------------------------------------------------------------------
// Input-gates GEMM: xg[dir][row][g] = x_bf[row'] . Wih[dir]^T + bias.
// 64x128 f32 tile per WG: 8 waves x 4 16x16 subtiles, K=192 = 6 k-steps.
// B staged N-major so fragment loads are contiguous b128s.
// Output rows stored sequence-major per direction (row = seq*48 + t).
// ---------------------------------------------------------------------------
__global__ __launch_bounds__(256) void gates_gemm(
    const unsigned short* __restrict__ xbf,   // [M_][C_]
    const unsigned short* __restrict__ wihT,  // [NGALL][C_]
    const float* __restrict__ biasP,          // [NGALL]
    float* __restrict__ xg)                   // [NDIR][M_][NG_]
{
  __shared__ unsigned short As[64][32];
  __shared__ unsigned short Bs[128][32];     // N-major tile
  const int tid = threadIdx.x, lane = tid & 31, wave = tid >> 5;
  const int nTile = blockIdx.x;              // 0..5 (128 cols each)
  const int mTile = blockIdx.y;              // 0..2303
  const bool vertical = nTile < 3;           // dirs 0,1 occupy cols [0,384)

  const int rLoc = tid >> 2, aChunk = tid & 3;   // A: 64 rows x 4 x uint4

  // Output row -> x row gather (vertical scan order is (b, w, h))
  int rg = mTile * 64 + rLoc, xrow;
  if (vertical) {
    int b = rg / SP_; int rem = rg - b * SP_; int w = rem / HH_; int h = rem - w * HH_;
    xrow = b * SP_ + h * WW_ + w;
  } else xrow = rg;

  const unsigned short* aSrc = xbf + (size_t)xrow * C_;
  const unsigned short* bSrc = wihT + (size_t)nTile * 128 * C_;
  const int mSub = wave & 3, nSubBase = (wave >> 2) * 4;

  v8f acc[4] = {};
  for (int ks = 0; ks < C_ / 32; ++ks) {
    const int k0 = ks * 32;
    *(uint4*)&As[rLoc][aChunk * 8] = *(const uint4*)(aSrc + k0 + aChunk * 8);
#pragma unroll
    for (int p = 0; p < 2; ++p) {            // B: 128 rows x 4 x uint4 = 2/thread
      int idx = tid + p * 256, row = idx >> 2, chunk = idx & 3;
      *(uint4*)&Bs[row][chunk * 8] = *(const uint4*)(bSrc + (size_t)row * C_ + k0 + chunk * 8);
    }
    __syncthreads();
    if (ks + 1 < C_ / 32) {
      __builtin_prefetch(aSrc + k0 + 32 + aChunk * 8, 0, 0);
      __builtin_prefetch(bSrc + (size_t)(tid >> 2) * C_ + k0 + 32 + (tid & 3) * 8, 0, 0);
    }
    v16bf a = load_a_frag(&As[mSub * 16][0], 32, lane);
#pragma unroll
    for (int j = 0; j < 4; ++j) {
      v16bf b = load_b_frag_t(&Bs[(nSubBase + j) * 16][0], 32, lane);
      acc[j] = __builtin_amdgcn_wmma_f32_16x16x32_bf16(false, a, false, b, (short)0, acc[j], false, false);
    }
    __syncthreads();
  }

  const int half = lane >> 4, nLane = lane & 15;
#pragma unroll
  for (int j = 0; j < 4; ++j) {
    int nGlob = nTile * 128 + (nSubBase + j) * 16 + nLane;
    int dir = nGlob / NG_;
    int gcol = nGlob - dir * NG_;
    size_t dirBase = (size_t)dir * M_ * NG_;
    float bv = biasP[nGlob];
#pragma unroll
    for (int r = 0; r < 8; ++r) {
      int rowOut = mTile * 64 + mSub * 16 + r + 8 * half;
      xg[dirBase + (size_t)rowOut * NG_ + gcol] = acc[j][r] + bv;
    }
  }
}

// ---------------------------------------------------------------------------
// Persistent recurrence: one WG owns 32 sequences of one direction.
// Whh^T fragments live in registers for all 48 steps (contiguous b128 loads
// from N-major storage); h state in LDS (bf16, K padded to 64); c in regs.
// Per step: stage xg_t coalesced into LDS, acc<-LDS, 2 WMMA k-steps/tile,
// scatter gates back, fused sigmoid/tanh pointwise, write bf16 h to LDS +
// the combined activation buffer in (b,h,w) order.
// ---------------------------------------------------------------------------
__global__ __launch_bounds__(256) void bilstm_scan(
    const float* __restrict__ xg,            // [NDIR][NSEQ*48][NG_], row = seq*48+t
    const unsigned short* __restrict__ whhT, // [NDIR][NG_][64]
    unsigned short* __restrict__ comb)       // [M_][NG_] bf16, col = dir*48 + u
{
  __shared__ unsigned short hB[32][64];
  __shared__ float gB[32][NG_];
  const int tid = threadIdx.x, lane = tid & 31, wave = tid >> 5;
  const int dir = blockIdx.y, seqBase = blockIdx.x * 32;
  const bool vertical = dir < 2; const int bwd = dir & 1;

  const int mSub = wave >> 2;                 // 24 tiles -> wave w owns 3w..3w+2
  const int nSub0 = (wave * 3) % 12;
  const int half = lane >> 4, nLane = lane & 15;

  const unsigned short* wdir = whhT + (size_t)dir * NG_ * 64;
  v16bf bfr[3][2];
#pragma unroll
  for (int j = 0; j < 3; ++j)
#pragma unroll
    for (int ks = 0; ks < 2; ++ks)
      bfr[j][ks] = load_b_frag_t(wdir + (size_t)(nSub0 + j) * 16 * 64 + ks * 32, 64, lane);

  for (int i = tid; i < 32 * 64; i += 256) (&hB[0][0])[i] = 0;
  float cst[6];
#pragma unroll
  for (int e = 0; e < 6; ++e) cst[e] = 0.0f;
  const int cell0 = tid * 6;                  // 32*48 cells / 256 threads
  const float* xgd = xg + (size_t)dir * M_ * NG_;
  const int sStage = tid >> 3, cStage = tid & 7;  // xg staging: 32 rows x 8 threads
  __syncthreads();

  for (int t = 0; t < HH_; ++t) {
    const int tt = bwd ? (HH_ - 1 - t) : t;
    // Coalesced stage of the 32x192 f32 gate block for timestep tt.
    {
      const float* src = xgd + ((size_t)(seqBase + sStage) * 48 + tt) * NG_;
#pragma unroll
      for (int q = 0; q < 6; ++q)
        *(float4*)&gB[sStage][(cStage * 6 + q) * 4] = *(const float4*)(src + (cStage * 6 + q) * 4);
    }
    __syncthreads();

    v8f acc[3];
#pragma unroll
    for (int j = 0; j < 3; ++j)
#pragma unroll
      for (int r = 0; r < 8; ++r)
        acc[j][r] = gB[mSub * 16 + r + 8 * half][(nSub0 + j) * 16 + nLane];

    v16bf a0 = load_a_frag(&hB[mSub * 16][0], 64, lane);
    v16bf a1 = load_a_frag(&hB[mSub * 16][32], 64, lane);
#pragma unroll
    for (int j = 0; j < 3; ++j) {
      acc[j] = __builtin_amdgcn_wmma_f32_16x16x32_bf16(false, a0, false, bfr[j][0], (short)0, acc[j], false, false);
      acc[j] = __builtin_amdgcn_wmma_f32_16x16x32_bf16(false, a1, false, bfr[j][1], (short)0, acc[j], false, false);
    }
#pragma unroll
    for (int j = 0; j < 3; ++j)
#pragma unroll
      for (int r = 0; r < 8; ++r)
        gB[mSub * 16 + r + 8 * half][(nSub0 + j) * 16 + nLane] = acc[j][r];
    __syncthreads();

#pragma unroll
    for (int e = 0; e < 6; ++e) {
      int cIdx = cell0 + e, s = cIdx / HD_, u = cIdx % HD_;
      float iv = gB[s][u], fv = gB[s][HD_ + u], gv = gB[s][2 * HD_ + u], ov = gB[s][3 * HD_ + u];
      float c = sigm(fv) * cst[e] + sigm(iv) * tanhf(gv);
      cst[e] = c;
      float h = sigm(ov) * tanhf(c);
      unsigned short hb = f2bf(h);
      hB[s][u] = hb;
      int seq = seqBase + s, b = seq / 48, q = seq - b * 48;
      int row = vertical ? (b * SP_ + tt * WW_ + q) : (b * SP_ + q * WW_ + tt);
      comb[(size_t)row * NG_ + dir * HD_ + u] = hb;
    }
    __syncthreads();   // protect hB/gB against next iteration
  }
}

// ---------------------------------------------------------------------------
// Projection: out[row][c] = comb[row][:] . proj_w[c][:] + proj_b[c]
// 64x64 tile, B staged N-major (proj_w natural layout).
// ---------------------------------------------------------------------------
__global__ __launch_bounds__(256) void proj_gemm(
    const unsigned short* __restrict__ abf,   // [M_][NG_]
    const unsigned short* __restrict__ projT, // [C_][NG_]
    const float* __restrict__ bias,           // [C_]
    float* __restrict__ out)                  // [M_][C_]
{
  __shared__ unsigned short As[64][32];
  __shared__ unsigned short Bs[64][32];
  const int tid = threadIdx.x, lane = tid & 31, wave = tid >> 5;
  const int nTile = blockIdx.x, mTile = blockIdx.y;
  const int rLoc = tid >> 2, chunk = tid & 3;
  const unsigned short* aSrc = abf + (size_t)(mTile * 64 + rLoc) * NG_;
  const unsigned short* bSrc = projT + (size_t)(nTile * 64 + rLoc) * NG_;
  const int mSub = wave & 3, nSubBase = (wave >> 2) * 2;

  v8f acc0 = {}, acc1 = {};
  for (int ks = 0; ks < NG_ / 32; ++ks) {
    const int k0 = ks * 32;
    *(uint4*)&As[rLoc][chunk * 8] = *(const uint4*)(aSrc + k0 + chunk * 8);
    *(uint4*)&Bs[rLoc][chunk * 8] = *(const uint4*)(bSrc + k0 + chunk * 8);
    __syncthreads();
    if (ks + 1 < NG_ / 32) {
      __builtin_prefetch(aSrc + k0 + 32 + chunk * 8, 0, 0);
      __builtin_prefetch(bSrc + k0 + 32 + chunk * 8, 0, 0);
    }
    v16bf a  = load_a_frag(&As[mSub * 16][0], 32, lane);
    v16bf b0 = load_b_frag_t(&Bs[nSubBase * 16][0], 32, lane);
    v16bf b1 = load_b_frag_t(&Bs[(nSubBase + 1) * 16][0], 32, lane);
    acc0 = __builtin_amdgcn_wmma_f32_16x16x32_bf16(false, a, false, b0, (short)0, acc0, false, false);
    acc1 = __builtin_amdgcn_wmma_f32_16x16x32_bf16(false, a, false, b1, (short)0, acc1, false, false);
    __syncthreads();
  }
  const int half = lane >> 4, nLane = lane & 15;
#pragma unroll
  for (int j = 0; j < 2; ++j) {
    const v8f acc = j ? acc1 : acc0;
    int n = nTile * 64 + (nSubBase + j) * 16 + nLane;
    float bv = bias[n];
#pragma unroll
    for (int r = 0; r < 8; ++r) {
      int row = mTile * 64 + mSub * 16 + r + 8 * half;
      out[(size_t)row * C_ + n] = acc[r] + bv;
    }
  }
}

extern "C" void kernel_launch(void* const* d_in, const int* in_sizes, int n_in,
                              void* d_out, int out_size, void* d_ws, size_t ws_size,
                              hipStream_t stream) {
  (void)in_sizes; (void)n_in; (void)out_size; (void)ws_size;
  const float* x    = (const float*)d_in[0];
  const float* vwf  = (const float*)d_in[1];
  const float* vhf  = (const float*)d_in[2];
  const float* vbif = (const float*)d_in[3];
  const float* vbhf = (const float*)d_in[4];
  const float* vwb  = (const float*)d_in[5];
  const float* vhb  = (const float*)d_in[6];
  const float* vbib = (const float*)d_in[7];
  const float* vbhb = (const float*)d_in[8];
  const float* hwf  = (const float*)d_in[9];
  const float* hhf  = (const float*)d_in[10];
  const float* hbif = (const float*)d_in[11];
  const float* hbhf = (const float*)d_in[12];
  const float* hwb  = (const float*)d_in[13];
  const float* hhb  = (const float*)d_in[14];
  const float* hbib = (const float*)d_in[15];
  const float* hbhb = (const float*)d_in[16];
  const float* projw = (const float*)d_in[17];
  const float* projb = (const float*)d_in[18];

  char* ws = (char*)d_ws;
  size_t off = 0;
  auto take = [&](size_t bytes) {
    size_t o = (off + 255) & ~(size_t)255; off = o + bytes; return (void*)(ws + o);
  };
  float*          xg    = (float*)take((size_t)NDIR * M_ * NG_ * sizeof(float));
  unsigned short* xbf   = (unsigned short*)take((size_t)M_ * C_ * sizeof(unsigned short));
  unsigned short* comb  = (unsigned short*)take((size_t)M_ * NG_ * sizeof(unsigned short));
  unsigned short* wihT  = (unsigned short*)take((size_t)NGALL * C_ * sizeof(unsigned short));
  unsigned short* whhT  = (unsigned short*)take((size_t)NDIR * NG_ * 64 * sizeof(unsigned short));
  unsigned short* projT = (unsigned short*)take((size_t)C_ * NG_ * sizeof(unsigned short));
  float*          biasP = (float*)take((size_t)NGALL * sizeof(float));

  pack_params<<<64, 256, 0, stream>>>(vwf, vhf, vbif, vbhf, vwb, vhb, vbib, vbhb,
                                      hwf, hhf, hbif, hbhf, hwb, hhb, hbib, hbhb,
                                      projw, wihT, whhT, projT, biasP);
  convert_x<<<2048, 256, 0, stream>>>(x, xbf, M_ * C_);
  gates_gemm<<<dim3(6, SP_), 256, 0, stream>>>(xbf, wihT, biasP, xg);
  bilstm_scan<<<dim3(NSEQ / 32, NDIR), 256, 0, stream>>>(xg, whhT, comb);
  proj_gemm<<<dim3(3, SP_), 256, 0, stream>>>(comb, projT, projb, (float*)d_out);
}
